// GAT_11879879544635
// MI455X (gfx1250) — matrix operand order repaired and checked by reading, not compiled, not run on hardware
//
#include <hip/hip_runtime.h>
#include <hip/hip_bf16.h>
#include <math.h>

typedef __attribute__((ext_vector_type(16))) __bf16 v16bf;
typedef __attribute__((ext_vector_type(8)))  __bf16 v8bf;
typedef __attribute__((ext_vector_type(8)))  float  v8f;

// ---------- helpers: order-preserving float<->uint for atomic max ----------
__device__ __forceinline__ unsigned fenc(float f) {
    unsigned u = __float_as_uint(f);
    return (u & 0x80000000u) ? ~u : (u | 0x80000000u);
}
__device__ __forceinline__ float fdec(unsigned u) {
    return (u & 0x80000000u) ? __uint_as_float(u & 0x7FFFFFFFu)
                             : __uint_as_float(~u);
}

// ---------- W (128x128 f32 row-major) -> Wt (bf16, Wt[col][k] = W[k][col]) ----------
__global__ void wt_kernel(const float* __restrict__ W, __bf16* __restrict__ Wt) {
    int t = blockIdx.x * blockDim.x + threadIdx.x;
    if (t >= 128 * 128) return;
    int k = t >> 7, j = t & 127;
    Wt[j * 128 + k] = (__bf16)W[k * 128 + j];
}

// ---------- GEMM: H[N x 128] = A[N x 128] (f32, cvt->bf16) * W (via Wt bf16) ----------
// one wave per 16x16 tile; block = 256 threads = 8 waves = all 8 column tiles of one row tile
__global__ void gemm_kernel(const float* __restrict__ A,
                            const __bf16* __restrict__ Wt,
                            float* __restrict__ H) {
    int wave = threadIdx.x >> 5;
    int lane = threadIdx.x & 31;
    int mt   = blockIdx.x;          // row tile (16 rows)
    int nt   = wave;                // col tile (16 cols), 0..7
    int half = lane >> 4;           // 0 for lanes 0-15, 1 for lanes 16-31
    int row  = mt * 16 + (lane & 15);
    int col  = nt * 16 + (lane & 15);

    v8f c = {};
#pragma unroll
    for (int kb = 0; kb < 128; kb += 32) {
        // A fragment: 16x32 bf16.  lane<16: K = kb..kb+15
        //                          lane>=16: K = kb+16..kb+31 (half*16)
        v16bf a;
        {
            const float* p0 = A + (size_t)row * 128 + kb + half * 16;
            const float* p1 = p0 + 8;
#pragma unroll
            for (int i = 0; i < 8; ++i) {
                a[i]     = (__bf16)p0[i];
                a[i + 8] = (__bf16)p1[i];
            }
        }
        // B fragment: 32x16 bf16, column = lane&15; lane<16: K=kb..kb+15, lane>=16: K=kb+16..kb+31
        v16bf b;
        {
            const v8bf* q = (const v8bf*)(Wt + (size_t)col * 128 + kb + half * 16);
            v8bf b0 = q[0], b1 = q[1];
#pragma unroll
            for (int i = 0; i < 8; ++i) { b[i] = b0[i]; b[i + 8] = b1[i]; }
        }
        c = __builtin_amdgcn_wmma_f32_16x16x32_bf16(false, a, false, b,
                                                    (short)0, c, false, false);
    }
    // C/D layout: VGPR r -> row (half*8 + r), col = lane&15
    int rbase = mt * 16 + half * 8;
#pragma unroll
    for (int r = 0; r < 8; ++r)
        H[(size_t)(rbase + r) * 128 + col] = c[r];
}

// ---------- per (node, head): attention logits halves + init stats + zero accumulator ----------
__global__ void prep_kernel(const float* __restrict__ H,
                            const float* __restrict__ a_src,
                            const float* __restrict__ a_dst,
                            float* __restrict__ al_s, float* __restrict__ al_d,
                            unsigned* __restrict__ m_u, float* __restrict__ s_arr,
                            float* __restrict__ acc, int Nn) {
    int t = blockIdx.x * blockDim.x + threadIdx.x;
    if (t >= Nn * 4) return;
    int n = t >> 2, hd = t & 3;
    const float* hp = H + (size_t)n * 128 + hd * 32;
    const float* as = a_src + hd * 32;
    const float* ad = a_dst + hd * 32;
    float ps = 0.f, pd = 0.f;
#pragma unroll
    for (int i = 0; i < 32; ++i) {
        float v = hp[i];
        ps += v * as[i];
        pd += v * ad[i];
    }
    al_s[t] = ps;
    al_d[t] = pd;
    m_u[t]  = 0u;    // 0 < fenc(-inf): identity for encoded max
    s_arr[t] = 0.f;
    float* ap = acc + (size_t)n * 128 + hd * 32;
#pragma unroll
    for (int i = 0; i < 32; ++i) ap[i] = 0.f;
}

// ---------- edge pass A: segment max of leaky_relu logits ----------
__global__ void edge_max_kernel(const int* __restrict__ ei,
                                const float* __restrict__ al_s,
                                const float* __restrict__ al_d,
                                unsigned* __restrict__ m_u, int E, int EP) {
    int t = blockIdx.x * blockDim.x + threadIdx.x;
    if (t >= EP) return;
    int s, d;
    if (t < E) { s = ei[t]; d = ei[E + t]; } else { s = d = t - E; }
#pragma unroll
    for (int hd = 0; hd < 4; ++hd) {
        float ev = al_s[s * 4 + hd] + al_d[d * 4 + hd];
        ev = ev < 0.f ? 0.2f * ev : ev;
        atomicMax(&m_u[d * 4 + hd], fenc(ev));
    }
}

// ---------- edge pass B: segment sum of exp(e - m) ----------
__global__ void edge_sum_kernel(const int* __restrict__ ei,
                                const float* __restrict__ al_s,
                                const float* __restrict__ al_d,
                                const unsigned* __restrict__ m_u,
                                float* __restrict__ s_arr, int E, int EP) {
    int t = blockIdx.x * blockDim.x + threadIdx.x;
    if (t >= EP) return;
    int s, d;
    if (t < E) { s = ei[t]; d = ei[E + t]; } else { s = d = t - E; }
#pragma unroll
    for (int hd = 0; hd < 4; ++hd) {
        float ev = al_s[s * 4 + hd] + al_d[d * 4 + hd];
        ev = ev < 0.f ? 0.2f * ev : ev;
        float m = fdec(m_u[d * 4 + hd]);
        atomicAdd(&s_arr[d * 4 + hd], __expf(ev - m));
    }
}

// ---------- edge pass C: acc[dst] += alpha * h[src]  (32 lanes x float4 per edge) ----------
__global__ void edge_msg_kernel(const int* __restrict__ ei,
                                const float* __restrict__ H,
                                const float* __restrict__ al_s,
                                const float* __restrict__ al_d,
                                const unsigned* __restrict__ m_u,
                                const float* __restrict__ s_arr,
                                float* __restrict__ acc, int E, int EP) {
    unsigned t = blockIdx.x * blockDim.x + threadIdx.x;
    unsigned e = t >> 5;
    int k = t & 31;                 // feature block: floats [4k, 4k+3]
    if (e >= (unsigned)EP) return;
    int s, d;
    if (e < (unsigned)E) { s = ei[e]; d = ei[E + e]; } else { s = d = (int)e - E; }
    int hd = k >> 3;                // head for this feature block
    float ev = al_s[s * 4 + hd] + al_d[d * 4 + hd];
    ev = ev < 0.f ? 0.2f * ev : ev;
    float m  = fdec(m_u[d * 4 + hd]);
    float sv = s_arr[d * 4 + hd];
    float alpha = __expf(ev - m) / (sv + 1e-16f);
    float4 hv = *(const float4*)(H + (size_t)s * 128 + k * 4);
    float* ap = acc + (size_t)d * 128 + k * 4;
    atomicAdd(ap + 0, hv.x * alpha);
    atomicAdd(ap + 1, hv.y * alpha);
    atomicAdd(ap + 2, hv.z * alpha);
    atomicAdd(ap + 3, hv.w * alpha);
}

// ---------- out = elu(acc + b), in place ----------
__global__ void bias_elu_kernel(float* __restrict__ acc,
                                const float* __restrict__ b, int Nn) {
    int t = blockIdx.x * blockDim.x + threadIdx.x;
    if (t >= Nn * 128) return;
    float v = acc[t] + b[t & 127];
    acc[t] = v > 0.f ? v : (__expf(v) - 1.0f);
}

// ---------- final FC 128->10 + log_softmax; one wave per node ----------
__global__ void fc_kernel(const float* __restrict__ X,
                          const float* __restrict__ Wf,
                          const float* __restrict__ bf,
                          float* __restrict__ out, int Nn) {
    int wave = threadIdx.x >> 5, lane = threadIdx.x & 31;
    int n = blockIdx.x * 8 + wave;
    if (n >= Nn) return;
    float4 xv = *(const float4*)(X + (size_t)n * 128 + lane * 4);
    float logit[10];
#pragma unroll
    for (int c = 0; c < 10; ++c) {
        int kb = lane * 4;
        float p = xv.x * Wf[(kb + 0) * 10 + c] + xv.y * Wf[(kb + 1) * 10 + c] +
                  xv.z * Wf[(kb + 2) * 10 + c] + xv.w * Wf[(kb + 3) * 10 + c];
#pragma unroll
        for (int off = 16; off > 0; off >>= 1) p += __shfl_xor(p, off, 32);
        logit[c] = p + bf[c];
    }
    if (lane == 0) {
        float mx = logit[0];
#pragma unroll
        for (int c = 1; c < 10; ++c) mx = fmaxf(mx, logit[c]);
        float se = 0.f;
#pragma unroll
        for (int c = 0; c < 10; ++c) se += __expf(logit[c] - mx);
        float lse = __logf(se);
#pragma unroll
        for (int c = 0; c < 10; ++c) out[(size_t)n * 10 + c] = logit[c] - mx - lse;
    }
}

extern "C" void kernel_launch(void* const* d_in, const int* in_sizes, int n_in,
                              void* d_out, int out_size, void* d_ws, size_t ws_size,
                              hipStream_t stream) {
    (void)n_in; (void)out_size; (void)ws_size;
    const float* x  = (const float*)d_in[0];
    const int*   ei = (const int*)d_in[1];
    int Nn = in_sizes[0] / 128;
    int E  = in_sizes[1] / 2;
    int EP = E + Nn;

    const float* Wl[3]  = {(const float*)d_in[2],  (const float*)d_in[6],  (const float*)d_in[10]};
    const float* asl[3] = {(const float*)d_in[3],  (const float*)d_in[7],  (const float*)d_in[11]};
    const float* adl[3] = {(const float*)d_in[4],  (const float*)d_in[8],  (const float*)d_in[12]};
    const float* bl[3]  = {(const float*)d_in[5],  (const float*)d_in[9],  (const float*)d_in[13]};
    const float* fcW = (const float*)d_in[14];
    const float* fcb = (const float*)d_in[15];

    char* ws = (char*)d_ws;
    float*    B0   = (float*)ws;    ws += (size_t)Nn * 128 * sizeof(float); // h
    float*    B1   = (float*)ws;    ws += (size_t)Nn * 128 * sizeof(float); // acc / next x
    float*    al_s = (float*)ws;    ws += (size_t)Nn * 4 * sizeof(float);
    float*    al_d = (float*)ws;    ws += (size_t)Nn * 4 * sizeof(float);
    unsigned* m_u  = (unsigned*)ws; ws += (size_t)Nn * 4 * sizeof(unsigned);
    float*    s_a  = (float*)ws;    ws += (size_t)Nn * 4 * sizeof(float);
    __bf16*   Wt   = (__bf16*)ws;   ws += (size_t)128 * 128 * sizeof(__bf16);

    const int TB = 256;
    int gN4   = (Nn * 4 + TB - 1) / TB;
    int gE    = (EP + TB - 1) / TB;
    int gMsg  = (int)(((long long)EP * 32 + TB - 1) / TB);
    int gFeat = (Nn * 128 + TB - 1) / TB;

    const float* Acur = x;
    for (int l = 0; l < 3; ++l) {
        wt_kernel<<<(128 * 128 + TB - 1) / TB, TB, 0, stream>>>(Wl[l], Wt);
        gemm_kernel<<<Nn / 16, TB, 0, stream>>>(Acur, Wt, B0);
        prep_kernel<<<gN4, TB, 0, stream>>>(B0, asl[l], adl[l], al_s, al_d, m_u, s_a, B1, Nn);
        edge_max_kernel<<<gE, TB, 0, stream>>>(ei, al_s, al_d, m_u, E, EP);
        edge_sum_kernel<<<gE, TB, 0, stream>>>(ei, al_s, al_d, m_u, s_a, E, EP);
        edge_msg_kernel<<<gMsg, TB, 0, stream>>>(ei, B0, al_s, al_d, m_u, s_a, B1, E, EP);
        bias_elu_kernel<<<gFeat, TB, 0, stream>>>(B1, bl[l], Nn);
        Acur = B1;
    }
    fc_kernel<<<(Nn + 7) / 8, TB, 0, stream>>>(B1, fcW, fcb, (float*)d_out, Nn);
}
